// DualGraphModel_2241972928706
// MI455X (gfx1250) — matrix operand order repaired and checked by reading, not compiled, not run on hardware
//
#include <hip/hip_runtime.h>

typedef _Float16 v16h __attribute__((ext_vector_type(16)));
typedef float    v8f  __attribute__((ext_vector_type(8)));
typedef unsigned int  u32;
typedef unsigned short u16;

// ---------------------------------------------------------------- helpers
__device__ __forceinline__ u32 pack2f16(float a, float b) {
    union { _Float16 h[2]; u32 u; } p;
    p.h[0] = (_Float16)a; p.h[1] = (_Float16)b;
    return p.u;
}

struct FragH { union { u32 u[8]; v16h v; }; };

// ---------------------------------------------------------------- degrees
__global__ void cnt_zero(int* cnt, int n) {
    int i = blockIdx.x * 256 + threadIdx.x;
    if (i < n) cnt[i] = 0;
}
__global__ void deg_count(const int* __restrict__ dst, int* __restrict__ cnt, int e) {
    int i = blockIdx.x * 256 + threadIdx.x;
    if (i < e) atomicAdd(&cnt[dst[i]], 1);
}
// dinv = rsqrt(in_degree + 1 self loop)
__global__ void dinv_from_cnt(const int* __restrict__ cnt, float* __restrict__ dinv, int n) {
    int i = blockIdx.x * 256 + threadIdx.x;
    if (i < n) dinv[i] = rsqrtf((float)(cnt[i] + 1));
}

// ------------------------------------------------- exclusive scan (1 block)
__global__ void exclusive_scan_1block(const int* __restrict__ cnt, u32* __restrict__ off, int n) {
    __shared__ u32 tmp[256];
    __shared__ u32 carry;
    const int tid = threadIdx.x;
    if (tid == 0) carry = 0;
    __syncthreads();
    for (int base = 0; base < n; base += 256) {
        u32 v = (base + tid < n) ? (u32)cnt[base + tid] : 0u;
        tmp[tid] = v;
        __syncthreads();
        #pragma unroll
        for (int o = 1; o < 256; o <<= 1) {
            u32 t = (tid >= o) ? tmp[tid - o] : 0u;
            __syncthreads();
            tmp[tid] += t;
            __syncthreads();
        }
        u32 c = carry;                     // all threads read carry
        if (base + tid < n) off[base + tid] = c + tmp[tid] - v;
        u32 total = tmp[255];
        __syncthreads();                   // all reads of carry/tmp done
        if (tid == 0) carry = c + total;
        __syncthreads();
    }
    if (tid == 0) off[n] = carry;
}

__global__ void cursor_init(const u32* __restrict__ off, u32* __restrict__ cur, int n) {
    int i = blockIdx.x * 256 + threadIdx.x;
    if (i < n) cur[i] = off[i];
}
__global__ void csr_fill(const int* __restrict__ src, const int* __restrict__ dst,
                         u32* __restrict__ cur, int* __restrict__ csr, int e) {
    int i = blockIdx.x * 256 + threadIdx.x;
    if (i < e) {
        u32 p = atomicAdd(&cur[dst[i]], 1u);
        csr[p] = src[i];
    }
}

// ------------------------------------------- weight convert + transpose (+pad)
// in:  f32 [K][nout] row-major     out: f16 [noutPad][K] row-major (Bt)
__global__ void convert_wt(const float* __restrict__ W, int K, int nout, int noutPad,
                           u16* __restrict__ out) {
    int idx = blockIdx.x * 256 + threadIdx.x;
    if (idx >= noutPad * K) return;
    int n = idx / K, k = idx - n * K;
    float v = (n < nout) ? W[(size_t)k * nout + n] : 0.f;
    union { _Float16 h; u16 s; } c; c.h = (_Float16)v;
    out[idx] = c.s;
}

// ---------------------------------------------------------------- WMMA GEMM
// C[M x nValid] (+colOff into ldc-pitched buffer) = A[M x K](f32) * Bt[BN x K](f16)^T
template<int BM, int BN, int WAVES_M, int WAVES_N, int TM, int TN>
__launch_bounds__(WAVES_M * WAVES_N * 32)
__global__ void gemm_wmma_f16(const float* __restrict__ A, int lda, int M, int K,
                              const u32* __restrict__ Btu,
                              float* __restrict__ C, int ldc, int colOff,
                              const float* __restrict__ bias, int nValid) {
    static_assert(BM == WAVES_M * TM * 16 && BN == WAVES_N * TN * 16, "tile mismatch");
    constexpr int NT  = WAVES_M * WAVES_N * 32;
    constexpr int LDU = 20;                       // 16 data dwords (BK=32 f16) + 4 pad

    __shared__ u32 As[BM * LDU];
    __shared__ u32 Bs[BN * LDU];

    const int tid   = threadIdx.x;
    const int lane  = tid & 31;
    const int wave  = tid >> 5;
    const int waveM = wave % WAVES_M;
    const int waveN = wave / WAVES_M;
    const int blockRow = blockIdx.x * BM;
    const int l16 = lane & 15;
    const int lhi = lane >> 4;

    v8f acc[TM][TN] = {};

    for (int k0 = 0; k0 < K; k0 += 32) {
        // ---- stage Bt tile via async copy to LDS (raw f16 bytes, ASYNCcnt)
        for (int lin = tid; lin < BN * 4; lin += NT) {
            int bn = lin >> 2, j = lin & 3;                       // 16B chunk j of row bn
            u32 lds = (u32)(size_t)(void*)&Bs[bn * LDU + j * 4];  // low 32 bits = LDS offset
            unsigned long long ga =
                (unsigned long long)(size_t)(Btu + (size_t)bn * (K >> 1) + (k0 >> 1) + j * 4);
            asm volatile("global_load_async_to_lds_b128 %0, %1, off"
                         :: "v"(lds), "v"(ga) : "memory");
        }
        // ---- stage A tile, converting f32 -> f16 on the fly
        for (int lin = tid; lin < BM * 8; lin += NT) {
            int row = lin >> 3, q = lin & 7;
            int gr = blockRow + row;
            float4 f = make_float4(0.f, 0.f, 0.f, 0.f);
            if (gr < M) f = *(const float4*)&A[(size_t)gr * lda + k0 + q * 4];
            As[row * LDU + q * 2]     = pack2f16(f.x, f.y);
            As[row * LDU + q * 2 + 1] = pack2f16(f.z, f.w);
        }
        asm volatile("s_wait_asynccnt 0" ::: "memory");
        __syncthreads();

        FragH fa[TM], fb[TN];
        #pragma unroll
        for (int tm = 0; tm < TM; ++tm) {
            int o = (waveM * TM * 16 + tm * 16 + l16) * LDU + (lhi << 2);
            *(uint4*)&fa[tm].u[0] = *(const uint4*)&As[o];      // K = kb .. kb+7
            *(uint4*)&fa[tm].u[4] = *(const uint4*)&As[o + 8];  // K = kb+16 .. kb+23
        }
        #pragma unroll
        for (int tn = 0; tn < TN; ++tn) {
            int o = (waveN * TN * 16 + tn * 16 + l16) * LDU + (lhi << 2);
            *(uint4*)&fb[tn].u[0] = *(const uint4*)&Bs[o];      // K = kb .. kb+7
            *(uint4*)&fb[tn].u[4] = *(const uint4*)&Bs[o + 8];  // K = kb+16 .. kb+23
        }
        #pragma unroll
        for (int tm = 0; tm < TM; ++tm)
            #pragma unroll
            for (int tn = 0; tn < TN; ++tn)
                acc[tm][tn] = __builtin_amdgcn_wmma_f32_16x16x32_f16(
                    false, fa[tm].v, false, fb[tn].v,
                    (short)0, acc[tm][tn], false, false);
        __syncthreads();
    }

    // epilogue: optional bias, masked store
    #pragma unroll
    for (int tm = 0; tm < TM; ++tm) {
        #pragma unroll
        for (int tn = 0; tn < TN; ++tn) {
            int row0 = blockRow + waveM * TM * 16 + tm * 16 + lhi * 8;
            int col  = waveN * TN * 16 + tn * 16 + l16;
            if (col < nValid) {
                float bv = bias ? bias[col] : 0.f;
                #pragma unroll
                for (int r = 0; r < 8; ++r) {
                    int m = row0 + r;
                    if (m < M) C[(size_t)m * ldc + colOff + col] = acc[tm][tn][r] + bv;
                }
            }
        }
    }
}

// ---------------- fused CSR gather + self-loop + bias + LN + ReLU + residual
// one wave32 per node; lane holds 4 of 128 channels in registers
__global__ void gather_ln_residual(const float* __restrict__ h, const int* __restrict__ csr,
                                   const u32* __restrict__ rowoff, const float* __restrict__ dinv,
                                   const float* __restrict__ b, const float* __restrict__ g,
                                   const float* __restrict__ be, float* __restrict__ xcat,
                                   int colOff, int n) {
    int node = blockIdx.x * 8 + (threadIdx.x >> 5);
    int lane = threadIdx.x & 31;
    if (node >= n) return;

    u32 beg = rowoff[node], end = rowoff[node + 1];   // wave-uniform
    float dn = dinv[node];

    const float4 hn = *(const float4*)&h[(size_t)node * 128 + lane * 4];
    float a0 = hn.x * dn, a1 = hn.y * dn, a2 = hn.z * dn, a3 = hn.w * dn;  // self loop (×dn later)

    for (u32 i = beg; i < end; ++i) {
        int s   = csr[i];                             // wave-uniform -> scalar load
        float w = dinv[s];
        const float4 hs = *(const float4*)&h[(size_t)s * 128 + lane * 4];
        a0 += hs.x * w; a1 += hs.y * w; a2 += hs.z * w; a3 += hs.w * w;
    }
    a0 *= dn; a1 *= dn; a2 *= dn; a3 *= dn;           // agg = dinv[d] * (sum + self)

    float4 bb = *(const float4*)&b[lane * 4];
    float v0 = a0 + bb.x, v1 = a1 + bb.y, v2 = a2 + bb.z, v3 = a3 + bb.w;

    float s1 = v0 + v1 + v2 + v3;
    #pragma unroll
    for (int o = 16; o > 0; o >>= 1) s1 += __shfl_xor(s1, o, 32);
    float mu = s1 * (1.f / 128.f);

    float d0 = v0 - mu, d1 = v1 - mu, d2 = v2 - mu, d3 = v3 - mu;
    float q = d0 * d0 + d1 * d1 + d2 * d2 + d3 * d3;
    #pragma unroll
    for (int o = 16; o > 0; o >>= 1) q += __shfl_xor(q, o, 32);
    float inv = rsqrtf(q * (1.f / 128.f) + 1e-5f);

    float4 gg = *(const float4*)&g[lane * 4];
    float4 ee = *(const float4*)&be[lane * 4];
    float y0 = fmaxf(d0 * inv * gg.x + ee.x, 0.f);
    float y1 = fmaxf(d1 * inv * gg.y + ee.y, 0.f);
    float y2 = fmaxf(d2 * inv * gg.z + ee.z, 0.f);
    float y3 = fmaxf(d3 * inv * gg.w + ee.w, 0.f);

    float* xp = &xcat[(size_t)node * 256 + colOff + lane * 4];
    float4 xv = *(const float4*)xp;
    xv.x += y0; xv.y += y1; xv.z += y2; xv.w += y3;
    *(float4*)xp = xv;
}

// ---------------------------------------------------------------- launch
extern "C" void kernel_launch(void* const* d_in, const int* in_sizes, int n_in,
                              void* d_out, int out_size, void* d_ws, size_t ws_size,
                              hipStream_t stream) {
    constexpr int H = 128, L = 4, IND = 256, CLS = 10;
    const int N = in_sizes[0] / IND;
    const int E = in_sizes[16] / 2;

    const float* x_r    = (const float*)d_in[0];
    const float* x_v    = (const float*)d_in[1];
    const float* Wred_r = (const float*)d_in[2];
    const float* bred_r = (const float*)d_in[3];
    const float* Wred_v = (const float*)d_in[4];
    const float* bred_v = (const float*)d_in[5];
    const float* W_r    = (const float*)d_in[6];
    const float* b_r    = (const float*)d_in[7];
    const float* g_r    = (const float*)d_in[8];
    const float* be_r   = (const float*)d_in[9];
    const float* W_v    = (const float*)d_in[10];
    const float* b_v    = (const float*)d_in[11];
    const float* g_v    = (const float*)d_in[12];
    const float* be_v   = (const float*)d_in[13];
    const float* Wcls   = (const float*)d_in[14];
    const float* bcls   = (const float*)d_in[15];
    const int*   ei_r   = (const int*)d_in[16];
    const int*   ei_v   = (const int*)d_in[17];
    const int *src_r = ei_r, *dst_r = ei_r + E;
    const int *src_v = ei_v, *dst_v = ei_v + E;
    float* out = (float*)d_out;

    // ---- workspace carve-up
    char* ws = (char*)d_ws;
    size_t off = 0;
    auto carve = [&](size_t bytes) -> char* {
        char* p = ws + off;
        off = (off + bytes + 255) & ~size_t(255);
        return p;
    };
    float* xcat     = (float*)carve((size_t)N * 256 * 4);  // r: cols 0..127, v: cols 128..255
    float* hbuf     = (float*)carve((size_t)N * H * 4);
    int*   cnt_r    = (int*)  carve((size_t)N * 4);
    int*   cnt_v    = (int*)  carve((size_t)N * 4);
    float* dinv_r   = (float*)carve((size_t)N * 4);
    float* dinv_v   = (float*)carve((size_t)N * 4);
    u32*   rowoff_r = (u32*)  carve((size_t)(N + 1) * 4);
    u32*   rowoff_v = (u32*)  carve((size_t)(N + 1) * 4);
    u32*   cursor_r = (u32*)  carve((size_t)N * 4);
    u32*   cursor_v = (u32*)  carve((size_t)N * 4);
    int*   csr_r    = (int*)  carve((size_t)E * 4);
    int*   csr_v    = (int*)  carve((size_t)E * 4);
    u16*   WredT_r  = (u16*)  carve((size_t)H * IND * 2);
    u16*   WredT_v  = (u16*)  carve((size_t)H * IND * 2);
    u16*   WT_r     = (u16*)  carve((size_t)L * H * H * 2);
    u16*   WT_v     = (u16*)  carve((size_t)L * H * H * 2);
    u16*   WclsT    = (u16*)  carve((size_t)16 * IND * 2);
    (void)ws_size; (void)n_in; (void)out_size;

    const int nbN = (N + 255) / 256;
    const int nbE = (E + 255) / 256;

    // ---- per-stream CSR build (amortized across all 4 layers)
    for (int s = 0; s < 2; ++s) {
        int*   cnt  = s ? cnt_v : cnt_r;
        float* dinv = s ? dinv_v : dinv_r;
        u32*   roff = s ? rowoff_v : rowoff_r;
        u32*   cur  = s ? cursor_v : cursor_r;
        int*   csr  = s ? csr_v : csr_r;
        const int* src = s ? src_v : src_r;
        const int* dst = s ? dst_v : dst_r;
        cnt_zero<<<nbN, 256, 0, stream>>>(cnt, N);
        deg_count<<<nbE, 256, 0, stream>>>(dst, cnt, E);
        dinv_from_cnt<<<nbN, 256, 0, stream>>>(cnt, dinv, N);
        exclusive_scan_1block<<<1, 256, 0, stream>>>(cnt, roff, N);
        cursor_init<<<nbN, 256, 0, stream>>>(roff, cur, N);
        csr_fill<<<nbE, 256, 0, stream>>>(src, dst, cur, csr, E);
    }

    // ---- weights -> f16, transposed
    convert_wt<<<(H * IND + 255) / 256, 256, 0, stream>>>(Wred_r, IND, H, H, WredT_r);
    convert_wt<<<(H * IND + 255) / 256, 256, 0, stream>>>(Wred_v, IND, H, H, WredT_v);
    for (int l = 0; l < L; ++l) {
        convert_wt<<<(H * H + 255) / 256, 256, 0, stream>>>(W_r + (size_t)l * H * H, H, H, H, WT_r + (size_t)l * H * H);
        convert_wt<<<(H * H + 255) / 256, 256, 0, stream>>>(W_v + (size_t)l * H * H, H, H, H, WT_v + (size_t)l * H * H);
    }
    convert_wt<<<(16 * IND + 255) / 256, 256, 0, stream>>>(Wcls, IND, CLS, 16, WclsT);

    const int gemmBlocks = (N + 127) / 128;

    // ---- dim reduction into xcat (bias fused)
    gemm_wmma_f16<128, 128, 4, 2, 2, 4><<<gemmBlocks, 256, 0, stream>>>(
        x_r, IND, N, IND, (const u32*)WredT_r, xcat, 256, 0, bred_r, H);
    gemm_wmma_f16<128, 128, 4, 2, 2, 4><<<gemmBlocks, 256, 0, stream>>>(
        x_v, IND, N, IND, (const u32*)WredT_v, xcat, 256, H, bred_v, H);

    // ---- GCN layers: WMMA GEMM + fused gather/LN/ReLU/residual
    for (int l = 0; l < L; ++l) {
        for (int s = 0; s < 2; ++s) {
            const int colOff   = s ? H : 0;
            const u16* WT      = s ? WT_v : WT_r;
            const float* dinv  = s ? dinv_v : dinv_r;
            const u32* roff    = s ? rowoff_v : rowoff_r;
            const int* csr     = s ? csr_v : csr_r;
            const float* bias  = (s ? b_v : b_r) + (size_t)l * H;
            const float* gamma = (s ? g_v : g_r) + (size_t)l * H;
            const float* beta  = (s ? be_v : be_r) + (size_t)l * H;

            gemm_wmma_f16<128, 128, 4, 2, 2, 4><<<gemmBlocks, 256, 0, stream>>>(
                xcat + colOff, 256, N, H, (const u32*)(WT + (size_t)l * H * H),
                hbuf, H, 0, nullptr, H);
            gather_ln_residual<<<(N + 7) / 8, 256, 0, stream>>>(
                hbuf, csr, roff, dinv, bias, gamma, beta, xcat, colOff, N);
        }
    }

    // ---- classifier: reads the implicit concat directly (K = 256)
    gemm_wmma_f16<128, 16, 8, 1, 1, 1><<<gemmBlocks, 256, 0, stream>>>(
        xcat, 256, N, IND, (const u32*)WclsT, out, CLS, 0, bcls, CLS);
}
